// PARGModel_73151882985680
// MI455X (gfx1250) — matrix operand order repaired and checked by reading, not compiled, not run on hardware
//
#include <hip/hip_runtime.h>
#include <math.h>

typedef __attribute__((ext_vector_type(16))) _Float16 v16h;
typedef __attribute__((ext_vector_type(8)))  float    v8f;
typedef __attribute__((ext_vector_type(2)))  __fp16   fp16x2;

union Frag16 { v16h h; unsigned u[8]; _Float16 e[16]; };

__device__ __forceinline__ float sigmoidf_(float x){ return 1.f/(1.f+__expf(-x)); }
// pack two f32 -> one dword of packed f16 (v_cvt_pk_rtz_f16_f32)
__device__ __forceinline__ unsigned pk2(float a, float b){
  union { fp16x2 h; unsigned u; } c;
  c.h = __builtin_amdgcn_cvt_pkrtz(a, b);
  return c.u;
}
// monotonic float<->uint encoding for atomicMax on floats
__device__ __forceinline__ unsigned encf(float f){
  unsigned u = __float_as_uint(f);
  return (u & 0x80000000u) ? ~u : (u | 0x80000000u);
}
__device__ __forceinline__ float decf(unsigned u){
  unsigned v = (u & 0x80000000u) ? (u & 0x7FFFFFFFu) : ~u;
  return __uint_as_float(v);
}

// ---------------------------------------------------------------------------
// Generic WMMA GEMM: C[M,N] = act( A[M,K] @ op(W) + bias ) (+ C if beta)
//   transW=1: W stored [N, ldw] row-major, use W^T  (torch Linear weight)
//   transW=0: W stored [K, ldw] row-major            (x @ W)
// Tile: 64x32 per 256-thread block (8 waves, each one 16x16 WMMA tile).
// Requires M%64==0 and N%32==0 (true for every call in this model);
// K may be arbitrary (zero-padded tail chunk takes the guarded path).
// LDS rows padded to 40 halves (80B) for bank-conflict-free b128 frag loads.
// ---------------------------------------------------------------------------
#define LDP 40
__global__ __launch_bounds__(256) void gemm_wmma(
    const float* __restrict__ A, int lda,
    const float* __restrict__ W, int ldw, int transW,
    const float* __restrict__ bias,
    float* __restrict__ C, int ldc,
    int M, int N, int K, int act, int beta)
{
  __shared__ _Float16 sA[64][LDP];   // [m][k]
  __shared__ _Float16 sB[32][LDP];   // [n][k]  (k pairs contiguous for B frag)
  int tid  = threadIdx.x;
  int wave = tid >> 5, lane = tid & 31;
  int half = lane >> 4, l15 = lane & 15;
  int m0 = blockIdx.x * 64;
  int n0 = blockIdx.y * 32;
  int wm = (wave >> 1) * 16;
  int wn = (wave & 1) * 16;

  v8f acc;
  #pragma unroll
  for (int r = 0; r < 8; ++r) acc[r] = 0.f;

  for (int k0 = 0; k0 < K; k0 += 32) {
    if (k0 + 32 <= K) {
      // ---- fast path: vectorized b128 loads, pk-convert, b128 LDS stores
      {
        int r  = tid >> 2;            // 64 rows, 4 threads/row
        int kq = (tid & 3) * 8;       // 8 k's per thread (16B aligned)
        const float* ap = A + (size_t)(m0 + r)*lda + k0 + kq;
        float4 a0 = *(const float4*)ap;
        float4 a1 = *(const float4*)(ap + 4);
        uint4 pk;
        pk.x = pk2(a0.x, a0.y); pk.y = pk2(a0.z, a0.w);
        pk.z = pk2(a1.x, a1.y); pk.w = pk2(a1.z, a1.w);
        *(uint4*)&sA[r][kq] = pk;
        if (k0 + 64 <= K) __builtin_prefetch(ap + 32, 0, 0);
      }
      if (transW) {
        int n  = tid >> 3;            // 32 rows, 8 threads/row
        int kq = (tid & 7) * 4;
        const float4 w0 = *(const float4*)(W + (size_t)(n0 + n)*ldw + k0 + kq);
        uint2 pk;
        pk.x = pk2(w0.x, w0.y); pk.y = pk2(w0.z, w0.w);
        *(uint2*)&sB[n][kq] = pk;
      } else {
        for (int i = tid; i < 32*32; i += 256) {
          int n = i >> 5, kk = i & 31;
          sB[n][kk] = (_Float16)W[(size_t)(k0 + kk)*ldw + n0 + n];
        }
      }
    } else {
      // ---- guarded tail (only the K=208 fusion GEMM hits this)
      for (int i = tid; i < 64*32; i += 256) {
        int r = i >> 5, kk = i & 31;
        int gk = k0 + kk;
        float v = (gk < K) ? A[(size_t)(m0 + r)*lda + gk] : 0.f;
        sA[r][kk] = (_Float16)v;
      }
      for (int i = tid; i < 32*32; i += 256) {
        int n = i >> 5, kk = i & 31;
        int gk = k0 + kk;
        float v = 0.f;
        if (gk < K)
          v = transW ? W[(size_t)(n0 + n)*ldw + gk] : W[(size_t)gk*ldw + n0 + n];
        sB[n][kk] = (_Float16)v;
      }
    }
    __syncthreads();

    Frag16 af, bfr;
    const unsigned* pa = (const unsigned*)&sA[wm + l15][0];
    #pragma unroll
    for (int i = 0; i < 4; ++i) {           // A: kk = half*8+2i and 16+half*8+2i
      af.u[i]     = pa[half*4 + i];
      af.u[4 + i] = pa[8 + half*4 + i];
    }
    const unsigned* pb = (const unsigned*)&sB[wn + l15][0];
    #pragma unroll
    for (int j = 0; j < 4; ++j) {           // B: kk = half*8+2j and 16+half*8+2j
      bfr.u[j]     = pb[half*4 + j];
      bfr.u[4 + j] = pb[8 + half*4 + j];
    }

    acc = __builtin_amdgcn_wmma_f32_16x16x32_f16(false, af.h, false, bfr.h,
                                                 (short)0, acc, false, false);
    __syncthreads();
  }

  int col = n0 + wn + l15;
  float bv = bias ? bias[col] : 0.f;
  #pragma unroll
  for (int r = 0; r < 8; ++r) {
    int row = m0 + wm + r + half*8;
    float v = acc[r] + bv;
    if (beta) v += C[(size_t)row*ldc + col];
    if (act == 1) v = v > 0.f ? v : 0.f;
    C[(size_t)row*ldc + col] = v;
  }
}

// ---------------------------------------------------------------------------
// Flash attention, one wave per (b, head, 16-query tile).  S=1024, D=128,
// H=4, dh=32.  qkv layout [B][S][384] (q|k|v).  out layout [B][S][128].
// ---------------------------------------------------------------------------
#define QKVW 384
__global__ __launch_bounds__(128) void attn_flash(
    const float* __restrict__ qkv, float* __restrict__ out)
{
  __shared__ _Float16 sP[4][16*LDP];   // per-wave P tile (C-layout -> A-layout)
  int tid  = threadIdx.x;
  int wave = tid >> 5, lane = tid & 31;
  int half = lane >> 4, l15 = lane & 15;
  int idx = blockIdx.x * 4 + wave;
  int qt = idx & 63;
  int h  = (idx >> 6) & 3;
  int b  = idx >> 8;

  const float scale = 0.17677669529663687f;   // 1/sqrt(32)
  Frag16 qf;
  {
    const float* qrow = qkv + ((size_t)(b*1024 + qt*16 + l15))*QKVW + h*32;
    float4 q0 = *(const float4*)(qrow + half*16);
    float4 q1 = *(const float4*)(qrow + half*16 + 4);
    float4 q2 = *(const float4*)(qrow + half*16 + 8);
    float4 q3 = *(const float4*)(qrow + half*16 + 12);
    qf.u[0] = pk2(q0.x*scale, q0.y*scale); qf.u[1] = pk2(q0.z*scale, q0.w*scale);
    qf.u[2] = pk2(q1.x*scale, q1.y*scale); qf.u[3] = pk2(q1.z*scale, q1.w*scale);
    qf.u[4] = pk2(q2.x*scale, q2.y*scale); qf.u[5] = pk2(q2.z*scale, q2.w*scale);
    qf.u[6] = pk2(q3.x*scale, q3.y*scale); qf.u[7] = pk2(q3.z*scale, q3.w*scale);
  }
  float m_[8], l_[8];
  v8f oc0, oc1;
  #pragma unroll
  for (int r = 0; r < 8; ++r){ m_[r] = -1e30f; l_[r] = 0.f; oc0[r]=0.f; oc1[r]=0.f; }

  for (int key0 = 0; key0 < 1024; key0 += 32) {
    Frag16 kf0, kf1;
    {
      const float* kr0 = qkv + ((size_t)(b*1024 + key0 + l15))*QKVW + 128 + h*32 + half*16;
      const float* kr1 = kr0 + (size_t)16*QKVW;
      float4 a0 = *(const float4*)(kr0);
      float4 a1 = *(const float4*)(kr0 + 4);
      float4 a2 = *(const float4*)(kr0 + 8);
      float4 a3 = *(const float4*)(kr0 + 12);
      kf0.u[0]=pk2(a0.x,a0.y); kf0.u[1]=pk2(a0.z,a0.w);
      kf0.u[2]=pk2(a1.x,a1.y); kf0.u[3]=pk2(a1.z,a1.w);
      kf0.u[4]=pk2(a2.x,a2.y); kf0.u[5]=pk2(a2.z,a2.w);
      kf0.u[6]=pk2(a3.x,a3.y); kf0.u[7]=pk2(a3.z,a3.w);
      float4 b0 = *(const float4*)(kr1);
      float4 b1 = *(const float4*)(kr1 + 4);
      float4 b2 = *(const float4*)(kr1 + 8);
      float4 b3 = *(const float4*)(kr1 + 12);
      kf1.u[0]=pk2(b0.x,b0.y); kf1.u[1]=pk2(b0.z,b0.w);
      kf1.u[2]=pk2(b1.x,b1.y); kf1.u[3]=pk2(b1.z,b1.w);
      kf1.u[4]=pk2(b2.x,b2.y); kf1.u[5]=pk2(b2.z,b2.w);
      kf1.u[6]=pk2(b3.x,b3.y); kf1.u[7]=pk2(b3.z,b3.w);
    }
    v8f z;
    #pragma unroll
    for (int r = 0; r < 8; ++r) z[r] = 0.f;
    v8f s0 = __builtin_amdgcn_wmma_f32_16x16x32_f16(false, qf.h, false, kf0.h, (short)0, z, false, false);
    v8f s1 = __builtin_amdgcn_wmma_f32_16x16x32_f16(false, qf.h, false, kf1.h, (short)0, z, false, false);

    _Float16* pt = &sP[wave][0];
    #pragma unroll
    for (int r = 0; r < 8; ++r) {           // online softmax per row
      float a = s0[r], c = s1[r];
      float mx = fmaxf(a, c);
      mx = fmaxf(mx, __shfl_xor(mx, 1, 32));
      mx = fmaxf(mx, __shfl_xor(mx, 2, 32));
      mx = fmaxf(mx, __shfl_xor(mx, 4, 32));
      mx = fmaxf(mx, __shfl_xor(mx, 8, 32));
      float mn   = fmaxf(m_[r], mx);
      float corr = __expf(m_[r] - mn);
      float e0 = __expf(a - mn);
      float e1 = __expf(c - mn);
      float rs = e0 + e1;
      rs += __shfl_xor(rs, 1, 32);
      rs += __shfl_xor(rs, 2, 32);
      rs += __shfl_xor(rs, 4, 32);
      rs += __shfl_xor(rs, 8, 32);
      l_[r] = l_[r]*corr + rs;
      m_[r] = mn;
      oc0[r] *= corr; oc1[r] *= corr;
      int row = r + 8*half;
      pt[row*LDP + l15]      = (_Float16)e0;
      pt[row*LDP + 16 + l15] = (_Float16)e1;
    }
    __syncthreads();

    Frag16 pf;
    {
      const unsigned* pp = (const unsigned*)&sP[wave][l15*LDP];
      #pragma unroll
      for (int i = 0; i < 4; ++i) {
        pf.u[i]     = pp[half*8 + i];
        pf.u[4 + i] = pp[half*8 + 4 + i];
      }
    }
    Frag16 vf0, vf1;
    {
      const float* vb = qkv + ((size_t)(b*1024 + key0))*QKVW + 256 + h*32;
      #pragma unroll
      for (int j = 0; j < 8; ++j) {
        int k2 = half*16 + 2*j;
        const float* r0 = vb + (size_t)k2*QKVW;
        const float* r1 = r0 + QKVW;
        vf0.u[j] = pk2(r0[l15],    r1[l15]);
        vf1.u[j] = pk2(r0[16+l15], r1[16+l15]);
      }
    }
    oc0 = __builtin_amdgcn_wmma_f32_16x16x32_f16(false, pf.h, false, vf0.h, (short)0, oc0, false, false);
    oc1 = __builtin_amdgcn_wmma_f32_16x16x32_f16(false, pf.h, false, vf1.h, (short)0, oc1, false, false);
    __syncthreads();
  }
  #pragma unroll
  for (int r = 0; r < 8; ++r) {
    float inv = 1.f / l_[r];
    int row = qt*16 + r + 8*half;
    float* orow = out + ((size_t)(b*1024 + row))*128 + h*32;
    orow[l15]    = oc0[r] * inv;
    orow[16+l15] = oc1[r] * inv;
  }
}

// ---------------------------------------------------------------------------
// Residual add + LayerNorm over D=128, one row per 128-thread block.
// ---------------------------------------------------------------------------
__global__ __launch_bounds__(128) void add_ln(
    const float* __restrict__ a, const float* __restrict__ b,
    const float* __restrict__ g, const float* __restrict__ bb,
    float* __restrict__ o)
{
  __shared__ float red[8];
  int row = blockIdx.x, t = threadIdx.x;
  int wave = t >> 5, lane = t & 31;
  size_t base = (size_t)row*128 + t;
  float v = a[base] + b[base];
  float s = v;
  s += __shfl_xor(s, 1, 32); s += __shfl_xor(s, 2, 32);
  s += __shfl_xor(s, 4, 32); s += __shfl_xor(s, 8, 32); s += __shfl_xor(s, 16, 32);
  if (lane == 0) red[wave] = s;
  __syncthreads();
  float mean = (red[0]+red[1]+red[2]+red[3]) * (1.f/128.f);
  float dv = v - mean;
  float s2 = dv*dv;
  s2 += __shfl_xor(s2, 1, 32); s2 += __shfl_xor(s2, 2, 32);
  s2 += __shfl_xor(s2, 4, 32); s2 += __shfl_xor(s2, 8, 32); s2 += __shfl_xor(s2, 16, 32);
  if (lane == 0) red[4+wave] = s2;
  __syncthreads();
  float var = (red[4]+red[5]+red[6]+red[7]) * (1.f/128.f);
  o[base] = dv * rsqrtf(var + 1e-5f) * g[t] + bb[t];
}

// ------------------------------- GAT kernels -------------------------------
__global__ void gat_scores(const float* __restrict__ xw,
                           const float* __restrict__ asrc, const float* __restrict__ adst,
                           float* __restrict__ as_, float* __restrict__ ad_,
                           int Nn, int heads, int chead)
{
  int i = blockIdx.x*blockDim.x + threadIdx.x;
  if (i >= Nn*heads) return;
  int n = i / heads, h = i % heads;
  const float* x = xw + (size_t)n*heads*chead + h*chead;
  const float* s = asrc + h*chead;
  const float* d = adst + h*chead;
  float accs = 0.f, accd = 0.f;
  for (int c = 0; c < chead; ++c) { float xv = x[c]; accs += xv*s[c]; accd += xv*d[c]; }
  as_[i] = accs; ad_[i] = accd;
}

__global__ void edge_logits(const int* __restrict__ src, const int* __restrict__ dst,
                            int Eo, int Nn, int heads,
                            const float* __restrict__ as_, const float* __restrict__ ad_,
                            float* __restrict__ elog, unsigned* __restrict__ menc)
{
  int i = blockIdx.x*blockDim.x + threadIdx.x;
  int ELh = (Eo + Nn) * heads;
  if (i >= ELh) return;
  int e = i / heads, h = i % heads;
  int s = e < Eo ? src[e] : e - Eo;
  int d = e < Eo ? dst[e] : e - Eo;
  float v = as_[s*heads + h] + ad_[d*heads + h];
  v = v > 0.f ? v : 0.2f*v;                 // leaky_relu 0.2
  elog[i] = v;
  atomicMax(&menc[d*heads + h], encf(v));
}

__global__ void edge_exp(const int* __restrict__ src, const int* __restrict__ dst,
                         int Eo, int Nn, int heads,
                         float* __restrict__ elog, const unsigned* __restrict__ menc,
                         float* __restrict__ den)
{
  int i = blockIdx.x*blockDim.x + threadIdx.x;
  int ELh = (Eo + Nn) * heads;
  if (i >= ELh) return;
  int e = i / heads, h = i % heads;
  int d = e < Eo ? dst[e] : e - Eo;
  float ex = __expf(elog[i] - decf(menc[d*heads + h]));
  elog[i] = ex;
  atomicAdd(&den[d*heads + h], ex);
}

__global__ void edge_aggr(const int* __restrict__ src, const int* __restrict__ dst,
                          int Eo, int Nn,
                          const float* __restrict__ ex, const float* __restrict__ den,
                          const float* __restrict__ xw, float* __restrict__ outp,
                          int heads, int chead)
{
  int e = blockIdx.x;
  int t = threadIdx.x;                     // heads*chead threads
  int s = (e < Eo) ? src[e] : (e - Eo);
  int d = (e < Eo) ? dst[e] : (e - Eo);
  int h = t / chead;
  float alpha = ex[(size_t)e*heads + h] / (den[(size_t)d*heads + h] + 1e-16f);
  atomicAdd(&outp[(size_t)d*heads*chead + t], alpha * xw[(size_t)s*heads*chead + t]);
}

__global__ void gat_post(const float* __restrict__ a, const float* __restrict__ bias,
                         float* __restrict__ o, int total, int C, int elu)
{
  int i = blockIdx.x*blockDim.x + threadIdx.x;
  if (i >= total) return;
  float v = a[i] + bias[i % C];
  if (elu) v = v > 0.f ? v : (__expf(v) - 1.f);
  o[i] = v;
}

// ------------------------- context / fusion / heads ------------------------
__global__ void ctx_mlp(const float* __restrict__ ctx, const float* __restrict__ Wc,
                        const float* __restrict__ bc, float* __restrict__ outc, int Bb)
{
  int i = blockIdx.x*blockDim.x + threadIdx.x;
  if (i >= Bb*16) return;
  int b = i / 16, j = i % 16;
  float acc = bc[j];
  for (int c = 0; c < 3; ++c) acc += ctx[b*3 + c] * Wc[j*3 + c];
  outc[i] = acc > 0.f ? acc : 0.f;
}

__global__ void fuse_k(const float* __restrict__ hsem, const float* __restrict__ t2,
                       const float* __restrict__ ctxh, const int* __restrict__ seq_lens,
                       const int* __restrict__ gb, float* __restrict__ fused,
                       int Nn, int Bb)
{
  int i = blockIdx.x*blockDim.x + threadIdx.x;
  if (i >= Nn*208) return;
  int k = i / 208, c = i % 208;
  float v;
  if (c < 128) {
    int csum = 0, samp = 0, start = 0;
    for (int b2 = 0; b2 < Bb; ++b2) {
      int l = seq_lens[b2];
      if (k < csum + l) { samp = b2; start = csum; break; }
      csum += l;
    }
    int off = k - start;
    v = hsem[((size_t)samp*1024 + off)*128 + c];
  } else if (c < 192) {
    v = t2[(size_t)k*64 + (c - 128)];
  } else {
    v = ctxh[gb[k]*16 + (c - 192)];
  }
  fused[(size_t)k*208 + c] = v;
}

__global__ void heads_k(const float* __restrict__ hf,
                        const float* __restrict__ Wr, const float* __restrict__ br,
                        const float* __restrict__ Wa, const float* __restrict__ ba,
                        const int* __restrict__ seq_lens,
                        float* __restrict__ outp, int Nn, int Bb)
{
  int k = blockIdx.x*blockDim.x + threadIdx.x;
  if (k < Nn) {
    float acc = ba[0];
    const float* hr = hf + (size_t)k*64;
    for (int c = 0; c < 64; ++c) acc += hr[c]*Wa[c];
    outp[Bb + k] = sigmoidf_(acc);
  }
  if (k < Bb) {
    int csum = 0;
    for (int i2 = 0; i2 <= k; ++i2) csum += seq_lens[i2];
    int tip = csum - 1;
    float acc = br[0];
    const float* hr = hf + (size_t)tip*64;
    for (int c = 0; c < 64; ++c) acc += hr[c]*Wr[c];
    outp[k] = sigmoidf_(acc);
  }
}

// ---------------------------------------------------------------------------
extern "C" void kernel_launch(void* const* d_in, const int* in_sizes, int n_in,
                              void* d_out, int out_size, void* d_ws, size_t ws_size,
                              hipStream_t stream)
{
  (void)n_in; (void)out_size; (void)ws_size;
  constexpr int Bc = 16, Sc = 1024, Dc = 128, DFFc = 2048;
  const int Mrows = Bc * Sc;
  const int Nn = in_sizes[2] / 128;
  const int Eo = in_sizes[3] / 2;
  const int EL = Eo + Nn;

  const float* seq_x      = (const float*)d_in[0];
  const int*   seq_lens   = (const int*)  d_in[1];
  const float* graph_x    = (const float*)d_in[2];
  const int*   edge_index = (const int*)  d_in[3];
  const float* context_x  = (const float*)d_in[4];
  const int*   graph_batch= (const int*)  d_in[5];
  // params in JAX pytree (sorted-key) order
  const float* Wa = (const float*)d_in[6];
  const float* Wc = (const float*)d_in[7];
  const float* Wf = (const float*)d_in[8];
  const float* Wr = (const float*)d_in[9];
  const float* ba = (const float*)d_in[10];
  const float* bc = (const float*)d_in[11];
  const float* bfv= (const float*)d_in[12];
  const float* br = (const float*)d_in[13];
  const float* g1W    = (const float*)d_in[14];
  const float* g1adst = (const float*)d_in[15];
  const float* g1asrc = (const float*)d_in[16];
  const float* g1b    = (const float*)d_in[17];
  const float* g2W    = (const float*)d_in[18];
  const float* g2adst = (const float*)d_in[19];
  const float* g2asrc = (const float*)d_in[20];
  const float* g2b    = (const float*)d_in[21];

  float* w = (float*)d_ws;
  auto alloc = [&](size_t n){ float* p = w; w += n; return p; };
  float* bufA  = alloc((size_t)Mrows*Dc);
  float* bufB  = alloc((size_t)Mrows*Dc);
  float* qkvb  = alloc((size_t)Mrows*3*Dc);
  float* attnb = alloc((size_t)Mrows*Dc);
  float* obuf  = alloc((size_t)Mrows*Dc);
  float* x1buf = alloc((size_t)Mrows*Dc);
  float* ffbuf = alloc((size_t)Mrows*Dc);
  float* midbuf= alloc((size_t)Mrows*512);
  float* xw1   = alloc((size_t)Nn*128);
  float* as1   = alloc((size_t)Nn*2);
  float* ad1   = alloc((size_t)Nn*2);
  float* elog1 = alloc((size_t)EL*2);
  unsigned* menc1 = (unsigned*)alloc((size_t)Nn*2);
  float* den1  = alloc((size_t)Nn*2);
  float* aggr1 = alloc((size_t)Nn*128);
  float* t1    = alloc((size_t)Nn*128);
  float* xw2   = alloc((size_t)Nn*64);
  float* as2   = alloc((size_t)Nn);
  float* ad2   = alloc((size_t)Nn);
  float* elog2 = alloc((size_t)EL);
  unsigned* menc2 = (unsigned*)alloc((size_t)Nn);
  float* den2  = alloc((size_t)Nn);
  float* aggr2 = alloc((size_t)Nn*64);
  float* t2    = alloc((size_t)Nn*64);
  float* ctxh  = alloc((size_t)Bc*16);
  float* fusedb= alloc((size_t)Nn*208);
  float* hfb   = alloc((size_t)Nn*64);

  // ---------------- Transformer encoder (2 layers) ----------------
  const float* hcur = seq_x;
  float* houts[2] = { bufA, bufB };
  for (int li = 0; li < 2; ++li) {
    int base = 22 + li*12;  // W1,W2,Wo,Wqkv,b1,b2,bf1,bf2,bo,bqkv,g1,g2
    const float* W1   = (const float*)d_in[base+0];
    const float* W2   = (const float*)d_in[base+1];
    const float* Wo   = (const float*)d_in[base+2];
    const float* Wqkv = (const float*)d_in[base+3];
    const float* b1   = (const float*)d_in[base+4];
    const float* b2   = (const float*)d_in[base+5];
    const float* bf1  = (const float*)d_in[base+6];
    const float* bf2  = (const float*)d_in[base+7];
    const float* bo   = (const float*)d_in[base+8];
    const float* bqkv = (const float*)d_in[base+9];
    const float* g1   = (const float*)d_in[base+10];
    const float* g2   = (const float*)d_in[base+11];

    gemm_wmma<<<dim3(Mrows/64, (3*Dc)/32), 256, 0, stream>>>(
        hcur, Dc, Wqkv, Dc, 1, bqkv, qkvb, 3*Dc, Mrows, 3*Dc, Dc, 0, 0);
    attn_flash<<<dim3(Bc*4*(Sc/16)/4), 128, 0, stream>>>(qkvb, attnb);
    gemm_wmma<<<dim3(Mrows/64, Dc/32), 256, 0, stream>>>(
        attnb, Dc, Wo, Dc, 1, bo, obuf, Dc, Mrows, Dc, Dc, 0, 0);
    add_ln<<<dim3(Mrows), 128, 0, stream>>>(hcur, obuf, g1, b1, x1buf);
    for (int ch = 0; ch < 4; ++ch) {   // FFN chunked over DFF (512 at a time)
      gemm_wmma<<<dim3(Mrows/64, 512/32), 256, 0, stream>>>(
          x1buf, Dc, W1 + (size_t)ch*512*Dc, Dc, 1, bf1 + ch*512,
          midbuf, 512, Mrows, 512, Dc, 1, 0);
      gemm_wmma<<<dim3(Mrows/64, Dc/32), 256, 0, stream>>>(
          midbuf, 512, W2 + ch*512, DFFc, 1, (ch==0 ? bf2 : nullptr),
          ffbuf, Dc, Mrows, Dc, 512, 0, (ch==0 ? 0 : 1));
    }
    add_ln<<<dim3(Mrows), 128, 0, stream>>>(x1buf, ffbuf, g2, b2, houts[li]);
    hcur = houts[li];
  }

  // ---------------- GAT stream ----------------
  (void)hipMemsetAsync(menc1, 0, (size_t)Nn*2*sizeof(unsigned), stream);
  (void)hipMemsetAsync(den1,  0, (size_t)Nn*2*sizeof(float),    stream);
  (void)hipMemsetAsync(aggr1, 0, (size_t)Nn*128*sizeof(float),  stream);
  (void)hipMemsetAsync(menc2, 0, (size_t)Nn*sizeof(unsigned),   stream);
  (void)hipMemsetAsync(den2,  0, (size_t)Nn*sizeof(float),      stream);
  (void)hipMemsetAsync(aggr2, 0, (size_t)Nn*64*sizeof(float),   stream);

  const int* esrc = edge_index;
  const int* edst = edge_index + Eo;

  gemm_wmma<<<dim3(Nn/64, 128/32), 256, 0, stream>>>(
      graph_x, 128, g1W, 128, 0, nullptr, xw1, 128, Nn, 128, 128, 0, 0);
  gat_scores<<<dim3((Nn*2+127)/128), 128, 0, stream>>>(xw1, g1asrc, g1adst, as1, ad1, Nn, 2, 64);
  edge_logits<<<dim3((EL*2+255)/256), 256, 0, stream>>>(esrc, edst, Eo, Nn, 2, as1, ad1, elog1, menc1);
  edge_exp   <<<dim3((EL*2+255)/256), 256, 0, stream>>>(esrc, edst, Eo, Nn, 2, elog1, menc1, den1);
  edge_aggr  <<<dim3(EL), 128, 0, stream>>>(esrc, edst, Eo, Nn, elog1, den1, xw1, aggr1, 2, 64);
  gat_post<<<dim3((Nn*128+255)/256), 256, 0, stream>>>(aggr1, g1b, t1, Nn*128, 128, 1);

  gemm_wmma<<<dim3(Nn/64, 64/32), 256, 0, stream>>>(
      t1, 128, g2W, 64, 0, nullptr, xw2, 64, Nn, 64, 128, 0, 0);
  gat_scores<<<dim3((Nn+127)/128), 128, 0, stream>>>(xw2, g2asrc, g2adst, as2, ad2, Nn, 1, 64);
  edge_logits<<<dim3((EL+255)/256), 256, 0, stream>>>(esrc, edst, Eo, Nn, 1, as2, ad2, elog2, menc2);
  edge_exp   <<<dim3((EL+255)/256), 256, 0, stream>>>(esrc, edst, Eo, Nn, 1, elog2, menc2, den2);
  edge_aggr  <<<dim3(EL), 64, 0, stream>>>(esrc, edst, Eo, Nn, elog2, den2, xw2, aggr2, 1, 64);
  gat_post<<<dim3((Nn*64+255)/256), 256, 0, stream>>>(aggr2, g2b, t2, Nn*64, 64, 0);

  // ---------------- context + fusion + heads ----------------
  ctx_mlp<<<dim3((Bc*16+127)/128), 128, 0, stream>>>(context_x, Wc, bc, ctxh, Bc);
  fuse_k<<<dim3((Nn*208+255)/256), 256, 0, stream>>>(hcur, t2, ctxh, seq_lens, graph_batch, fusedb, Nn, Bc);
  gemm_wmma<<<dim3(Nn/64, 64/32), 256, 0, stream>>>(
      fusedb, 208, Wf, 208, 1, bfv, hfb, 64, Nn, 64, 208, 1, 0);
  heads_k<<<dim3((Nn+255)/256), 256, 0, stream>>>(hfb, Wr, br, Wa, ba, seq_lens,
                                                  (float*)d_out, Nn, Bc);
}
